// MLABlock_86002425135556
// MI455X (gfx1250) — matrix operand order repaired and checked by reading, not compiled, not run on hardware
//
#include <hip/hip_runtime.h>

// ---------------------------------------------------------------------------
// Problem constants (match reference)
// ---------------------------------------------------------------------------
constexpr int cB  = 2;
constexpr int cS  = 1024;
constexpr int cH  = 1024;
constexpr int cNH = 16;
constexpr int cHD = 64;
constexpr int cW  = 4;
constexpr int cKS = (cW + 1) * cS;       // 5120 keys per (b,h)
constexpr float cEPS = 1e-5f;

typedef __attribute__((ext_vector_type(8)))  _Float16 half8;
typedef __attribute__((ext_vector_type(16))) _Float16 v16h;
typedef __attribute__((ext_vector_type(8)))  float    v8f;
typedef __attribute__((ext_vector_type(4)))  unsigned int u32x4;
typedef __attribute__((ext_vector_type(8)))  int      i32x8;
typedef __attribute__((ext_vector_type(4)))  int      i32x4;

static __device__ __forceinline__ v16h h8cat(half8 lo, half8 hi) {
    v16h r;
#pragma unroll
    for (int i = 0; i < 8; ++i) { r[i] = lo[i]; r[i + 8] = hi[i]; }
    return r;
}

static __device__ __forceinline__ v8f wmma32(v16h a, v16h b, v8f c) {
    // D = A(16x32 f16) * B(32x16 f16) + C(16x16 f32)
    return __builtin_amdgcn_wmma_f32_16x16x32_f16(
        /*neg_a=*/false, a, /*neg_b=*/false, b,
        /*c_mod=*/(short)0, c, /*reuse_a=*/false, /*reuse_b=*/false);
}

// ---------------------------------------------------------------------------
// Tensor Data Mover helpers (CDNA5).  2D tile load: tile_d0 elements (f16)
// along dim0 x tile_d1 rows, row stride = d0_stride elements.
// D# bit packing per CDNA5 ISA.
// ---------------------------------------------------------------------------
static __device__ __forceinline__ void tdm_load_2d(unsigned int lds_off,
                                                   unsigned long long gaddr,
                                                   unsigned int tensor_d0,
                                                   unsigned int tensor_d1,
                                                   unsigned int d0_stride,
                                                   unsigned int tile_d0,
                                                   unsigned int tile_d1) {
#if __has_builtin(__builtin_amdgcn_tensor_load_to_lds)
    u32x4 g0;
    g0[0] = 1u;                                           // count=1, user D#
    g0[1] = lds_off;                                      // lds_addr (bytes)
    g0[2] = (unsigned int)(gaddr & 0xffffffffu);          // global_addr[31:0]
    g0[3] = (unsigned int)((gaddr >> 32) & 0x01ffffffu)   // global_addr[56:32]
            | (2u << 30);                                 // type = 2 (image)
    i32x8 g1;
    g1[0] = (int)(1u << 16);                              // data_size=1 -> 2B
    g1[1] = (int)((tensor_d0 & 0xffffu) << 16);           // tensor_dim0[15:0]
    g1[2] = (int)((tensor_d0 >> 16) | ((tensor_d1 & 0xffffu) << 16));
    g1[3] = (int)((tensor_d1 >> 16) | (tile_d0 << 16));   // tile_dim0
    g1[4] = (int)(tile_d1 & 0xffffu);                     // tile_dim1 (dim2=0)
    g1[5] = (int)d0_stride;                               // dim0_stride[31:0]
    g1[6] = 0;                                            // stride hi, dim1_stride lo
    g1[7] = 0;
    i32x4 z4 = {0, 0, 0, 0};
#if __clang_major__ >= 23
    i32x8 z8 = {};
    __builtin_amdgcn_tensor_load_to_lds(g0, g1, z4, z4, z8, 0);
#else
    __builtin_amdgcn_tensor_load_to_lds(g0, g1, z4, z4, 0);
#endif
#else
    (void)lds_off; (void)gaddr; (void)tensor_d0; (void)tensor_d1;
    (void)d0_stride; (void)tile_d0; (void)tile_d1;
#endif
}

static __device__ __forceinline__ void tdm_wait0() {
#if __has_builtin(__builtin_amdgcn_s_wait_tensorcnt)
    __builtin_amdgcn_s_wait_tensorcnt((short)0);
#endif
}

// Truncating a flat LDS pointer to 32 bits yields the LDS byte offset
// (LDS aperture lives in addr[63:32]).
static __device__ __forceinline__ unsigned int lds_off_u32(const void* p) {
    return (unsigned int)(unsigned long long)(uintptr_t)p;
}

// 256-thread block reduction (sum)
static __device__ __forceinline__ float block_sum256(float v, float* sh) {
    int t = threadIdx.x;
    sh[t] = v;
    __syncthreads();
#pragma unroll
    for (int off = 128; off > 0; off >>= 1) {
        if (t < off) sh[t] += sh[t + off];
        __syncthreads();
    }
    float r = sh[0];
    __syncthreads();
    return r;
}

// ---------------------------------------------------------------------------
// 1) LayerNorm(hidden) -> f16 rows of kv buffer at [b][4S + s]
// ---------------------------------------------------------------------------
__global__ void ln_hidden_kernel(const float* __restrict__ x,
                                 const float* __restrict__ scale,
                                 const float* __restrict__ bias,
                                 _Float16* __restrict__ kv) {
    __shared__ float sh[256];
    int row = blockIdx.x;                 // b*S + s
    int b = row >> 10, s = row & 1023;
    const float* xr = x + (size_t)row * cH;

    float ls = 0.f;
    for (int i = threadIdx.x; i < cH; i += 256) ls += xr[i];
    float mu = block_sum256(ls, sh) * (1.0f / cH);

    float lv = 0.f;
    for (int i = threadIdx.x; i < cH; i += 256) {
        float d = xr[i] - mu; lv += d * d;
    }
    float var = block_sum256(lv, sh) * (1.0f / cH);
    float inv = rsqrtf(var + cEPS);

    _Float16* o = kv + ((size_t)b * cKS + (size_t)cW * cS + s) * cH;
    for (int i = threadIdx.x; i < cH; i += 256)
        o[i] = (_Float16)((xr[i] - mu) * inv * scale[i] + bias[i]);
}

// ---------------------------------------------------------------------------
// 2) cache f32 [W,B,S,H] -> kv f16 rows [b][w*S + s]
// ---------------------------------------------------------------------------
__global__ void cache_to_kv_kernel(const float* __restrict__ cache,
                                   _Float16* __restrict__ kv) {
    size_t i = (size_t)blockIdx.x * blockDim.x + threadIdx.x;  // W*B*S*H elems
    int h0 = (int)(i & (cH - 1));
    size_t r = i >> 10;                 // (w*B + b)*S + s
    int s  = (int)(r & (cS - 1));
    int wb = (int)(r >> 10);
    int b  = wb & (cB - 1);
    int w  = wb >> 1;
    kv[((size_t)b * cKS + (size_t)w * cS + s) * cH + h0] = (_Float16)cache[i];
}

// ---------------------------------------------------------------------------
// 3) Weight fp32 [K,N] -> f16 transposed [N,K]
// ---------------------------------------------------------------------------
__global__ void wt_convert_kernel(const float* __restrict__ Win,
                                  _Float16* __restrict__ WT) {
    size_t i = (size_t)blockIdx.x * blockDim.x + threadIdx.x;  // N*K elems
    int k = (int)(i & (cH - 1));
    int n = (int)(i >> 10);
    WT[i] = (_Float16)Win[(size_t)k * cH + n];
}

// ---------------------------------------------------------------------------
// 4) TDM-staged WMMA GEMM:  C[M,N] f32 = A[M,K] f16 x BT[N,K] f16.
//    Block = 8 waves, 128x128 output tile, K chunks of 64, double-buffered
//    LDS fed by tensor_load_to_lds; waves consume ds_load_b128 fragments.
//    A-row remap (rowsPerBatch/batchStride/baseOff) applied per 128-row tile
//    (tiles never straddle a batch boundary: rowsPerBatch % 128 == 0).
// ---------------------------------------------------------------------------
__global__ void gemm_tdm_kernel(const _Float16* __restrict__ A,
                                const _Float16* __restrict__ BT,
                                float* __restrict__ C,
                                int M, int N, int K,
                                int rowsPerBatch, int batchStride, int baseOff) {
    __shared__ _Float16 Abuf[2][128][64];   // 32 KB
    __shared__ _Float16 Bbuf[2][128][64];   // 32 KB
    int lane = threadIdx.x & 31;
    int wv   = threadIdx.x >> 5;
    int l16 = lane & 15, hid = lane >> 4;

    int mtiles = M >> 7;
    int m0 = (blockIdx.x % mtiles) * 128;
    int n0 = (blockIdx.x / mtiles) * 128;
    int arow0 = (m0 / rowsPerBatch) * batchStride + baseOff + (m0 % rowsPerBatch);

    int wm = (wv & 3) * 32;                 // wave covers rows [wm, wm+32)
    int wn = (wv >> 2) * 64;                // wave covers cols [wn, wn+64)

    v8f acc[2][4] = {};

    const int nchunk = K >> 6;              // K/64
    if (wv == 0) {
        tdm_load_2d(lds_off_u32(&Abuf[0][0][0]),
                    (unsigned long long)(uintptr_t)(A + (size_t)arow0 * K),
                    (unsigned)K, 1u << 20, (unsigned)K, 64u, 128u);
        tdm_load_2d(lds_off_u32(&Bbuf[0][0][0]),
                    (unsigned long long)(uintptr_t)(BT + (size_t)n0 * K),
                    (unsigned)K, 1u << 20, (unsigned)K, 64u, 128u);
        tdm_wait0();
    }
    __syncthreads();

    for (int kc = 0; kc < nchunk; ++kc) {
        int cur = kc & 1, nxt = cur ^ 1;
        if (wv == 0 && kc + 1 < nchunk) {
            int k = (kc + 1) * 64;
            tdm_load_2d(lds_off_u32(&Abuf[nxt][0][0]),
                        (unsigned long long)(uintptr_t)(A + (size_t)arow0 * K + k),
                        (unsigned)K, 1u << 20, (unsigned)K, 64u, 128u);
            tdm_load_2d(lds_off_u32(&Bbuf[nxt][0][0]),
                        (unsigned long long)(uintptr_t)(BT + (size_t)n0 * K + k),
                        (unsigned)K, 1u << 20, (unsigned)K, 64u, 128u);
        }
        // 16 WMMA on the current buffer
#pragma unroll
        for (int kk = 0; kk < 2; ++kk) {
            v16h af[2];
#pragma unroll
            for (int mt = 0; mt < 2; ++mt) {
                const half8* p0 = (const half8*)&Abuf[cur][wm + mt * 16 + l16][kk * 32 + hid * 8];
                const half8* p1 = (const half8*)&Abuf[cur][wm + mt * 16 + l16][kk * 32 + 16 + hid * 8];
                af[mt] = h8cat(*p0, *p1);
            }
#pragma unroll
            for (int nt = 0; nt < 4; ++nt) {
                const half8* q0 = (const half8*)&Bbuf[cur][wn + nt * 16 + l16][kk * 32 + hid * 16];
                v16h bf = h8cat(q0[-hid], q0[2 - hid]);
#pragma unroll
                for (int mt = 0; mt < 2; ++mt)
                    acc[mt][nt] = wmma32(af[mt], bf, acc[mt][nt]);
            }
        }
        if (wv == 0 && kc + 1 < nchunk) tdm_wait0();
        __syncthreads();
    }

#pragma unroll
    for (int mt = 0; mt < 2; ++mt)
#pragma unroll
        for (int nt = 0; nt < 4; ++nt)
#pragma unroll
            for (int j = 0; j < 8; ++j)
                C[(size_t)(m0 + wm + mt * 16 + hid * 8 + j) * N
                  + (n0 + wn + nt * 16 + l16)] = acc[mt][nt][j];
}

// ---------------------------------------------------------------------------
// 5) RoPE on Q: qout f32 [B*S, NH*HD] -> Qf16 [B][NH][S][HD]
// ---------------------------------------------------------------------------
__global__ void rope_q_kernel(const float* __restrict__ qo,
                              _Float16* __restrict__ Qf) {
    size_t i = (size_t)blockIdx.x * blockDim.x + threadIdx.x;  // B*S*NH*32
    int d = (int)(i & 31);
    size_t t = i >> 5;
    int h = (int)(t & (cNH - 1));
    int m = (int)(t >> 4);                 // b*S + s
    int s = m & (cS - 1);
    int b = m >> 10;
    float x1 = qo[(size_t)m * cH + h * cHD + d];
    float x2 = qo[(size_t)m * cH + h * cHD + d + 32];
    float fr = __expf(-(float)d * 0.2878231366f);  // ln(10000)/32
    float ang = (float)s * fr;
    float cs = __cosf(ang), sn = __sinf(ang);
    size_t o = ((size_t)(b * cNH + h) * cS + s) * cHD;
    Qf[o + d]      = (_Float16)(x1 * cs - x2 * sn);
    Qf[o + d + 32] = (_Float16)(x1 * sn + x2 * cs);
}

// RoPE on K: kout f32 [B*KS, NH*HD] -> Kf16 [B][NH][KS][HD]
__global__ void rope_k_kernel(const float* __restrict__ ko,
                              _Float16* __restrict__ Kf) {
    size_t i = (size_t)blockIdx.x * blockDim.x + threadIdx.x;  // B*KS*NH*32
    int d = (int)(i & 31);
    size_t t = i >> 5;
    int h = (int)(t & (cNH - 1));
    int m = (int)(t >> 4);                 // b*KS + r
    int r = m % cKS;
    int b = m / cKS;
    int pos = r & (cS - 1);
    float x1 = ko[(size_t)m * cH + h * cHD + d];
    float x2 = ko[(size_t)m * cH + h * cHD + d + 32];
    float fr = __expf(-(float)d * 0.2878231366f);
    float ang = (float)pos * fr;
    float cs = __cosf(ang), sn = __sinf(ang);
    size_t o = ((size_t)(b * cNH + h) * cKS + r) * cHD;
    Kf[o + d]      = (_Float16)(x1 * cs - x2 * sn);
    Kf[o + d + 32] = (_Float16)(x1 * sn + x2 * cs);
}

// V transpose: vout f32 [B*KS, NH*HD] -> Vt f16 [B][NH][HD][KS]
__global__ void v_transpose_kernel(const float* __restrict__ vo,
                                   _Float16* __restrict__ Vt) {
    size_t i = (size_t)blockIdx.x * blockDim.x + threadIdx.x;  // B*KS*NH*HD
    int col = (int)(i & (cH - 1));
    int m   = (int)(i >> 10);              // b*KS + r
    int h = col >> 6, d = col & 63;
    int r = m % cKS;
    int b = m / cKS;
    Vt[((size_t)(b * cNH + h) * cHD + d) * cKS + r] = (_Float16)vo[i];
}

// ---------------------------------------------------------------------------
// 6) Flash attention: one wave per 16-query tile per (b,h).
//    scores: Q(16x64) x K^T(64x16) via 2 wmma/k-step, online softmax,
//    P(16x32) x V(32x64) via 4 wmma, LDS round-trip relayouts P into A-frag.
// ---------------------------------------------------------------------------
__global__ void attn_kernel(const _Float16* __restrict__ Qf,
                            const _Float16* __restrict__ Kf,
                            const _Float16* __restrict__ Vt,
                            _Float16* __restrict__ ctx) {
    __shared__ _Float16 lds[8][16][32];     // per-wave P tile (16 q x 32 k)
    int lane = threadIdx.x & 31;
    int wv   = threadIdx.x >> 5;
    int wid  = blockIdx.x * 8 + wv;
    int qt = wid & 63;                      // 64 q-tiles per head
    int h  = (wid >> 6) & (cNH - 1);
    int b  = wid >> 10;
    int l16 = lane & 15, hid = lane >> 4;

    const _Float16* Qb = Qf + ((size_t)(b * cNH + h) * cS + qt * 16) * cHD;
    const _Float16* Kb = Kf + (size_t)(b * cNH + h) * cKS * cHD;
    const _Float16* Vb = Vt + (size_t)(b * cNH + h) * cHD * cKS;

    // Q A-fragments for k-dim d in [0,32) and [32,64)
    v16h qa[2];
#pragma unroll
    for (int t = 0; t < 2; ++t) {
        const half8* p0 = (const half8*)(Qb + (size_t)l16 * cHD + t * 32 + hid * 8);
        const half8* p1 = (const half8*)(Qb + (size_t)l16 * cHD + t * 32 + 16 + hid * 8);
        qa[t] = h8cat(*p0, *p1);
    }

    float mrow[8], lrow[8];
    v8f acc[4] = {};
#pragma unroll
    for (int j = 0; j < 8; ++j) { mrow[j] = -1e30f; lrow[j] = 0.f; }
    const float scale = 0.125f;             // 1/sqrt(64)

    for (int kb = 0; kb < cKS; kb += 32) {
        v8f sfr[2] = {};
#pragma unroll
        for (int t = 0; t < 2; ++t) {       // two 16-key sub-tiles
            int kcol = kb + t * 16 + l16;
#pragma unroll
            for (int dk = 0; dk < 2; ++dk) {
                const half8* p0 = (const half8*)(Kb + (size_t)kcol * cHD + dk * 32 + hid * 16);
                v16h bf = h8cat(p0[-hid], p0[2 - hid]);
                sfr[t] = wmma32(qa[dk], bf, sfr[t]);
            }
        }
        // --- online softmax: per-row max over 32 new keys + running max ---
        float mnew[8];
#pragma unroll
        for (int j = 0; j < 8; ++j) {
            sfr[0][j] *= scale; sfr[1][j] *= scale;
            float t = fmaxf(sfr[0][j], sfr[1][j]);
#pragma unroll
            for (int msk = 8; msk > 0; msk >>= 1)
                t = fmaxf(t, __shfl_xor(t, msk, 32));
            mnew[j] = fmaxf(mrow[j], t);
        }
#pragma unroll
        for (int j = 0; j < 8; ++j) {
            float f = __expf(mrow[j] - mnew[j]);
            mrow[j] = mnew[j];
            float p0 = __expf(sfr[0][j] - mnew[j]);
            float p1 = __expf(sfr[1][j] - mnew[j]);
            int r = hid * 8 + j;            // C-frag row
            lds[wv][r][l16]      = (_Float16)p0;
            lds[wv][r][16 + l16] = (_Float16)p1;
            float rs = p0 + p1;
#pragma unroll
            for (int msk = 8; msk > 0; msk >>= 1)
                rs += __shfl_xor(rs, msk, 32);
            lrow[j] = lrow[j] * f + rs;
#pragma unroll
            for (int d = 0; d < 4; ++d) acc[d][j] *= f;
        }
        // --- P (A-frag) from LDS, then P x V ---
        const half8* lp0 = (const half8*)&lds[wv][l16][hid * 8];
        const half8* lp1 = (const half8*)&lds[wv][l16][16 + hid * 8];
        v16h pa = h8cat(*lp0, *lp1);
#pragma unroll
        for (int d = 0; d < 4; ++d) {
            int dcol = d * 16 + l16;
            const half8* p0 = (const half8*)(Vb + (size_t)dcol * cKS + kb + hid * 16);
            v16h vb = h8cat(p0[-hid], p0[2 - hid]);
            acc[d] = wmma32(pa, vb, acc[d]);
        }
    }

    // normalize and store ctx as f16 rows [B*S][NH*HD]
    _Float16* Cb = ctx + ((size_t)(b * cS + qt * 16)) * (cNH * cHD) + h * cHD;
#pragma unroll
    for (int j = 0; j < 8; ++j) {
        float inv = 1.0f / lrow[j];
        int r = hid * 8 + j;
#pragma unroll
        for (int d = 0; d < 4; ++d)
            Cb[(size_t)r * (cNH * cHD) + d * 16 + l16] = (_Float16)(acc[d][j] * inv);
    }
}

// ---------------------------------------------------------------------------
// 7) Epilogue: gated residual + layer scale -> output; LayerNorm(output)
//    -> new cache slot (W-1).  One block per (b,s) row.
// ---------------------------------------------------------------------------
__global__ void epilogue_kernel(const float* __restrict__ hidden,
                                const float* __restrict__ cache,
                                const float* __restrict__ attn_out,
                                const float* __restrict__ gate,
                                const float* __restrict__ lscale,
                                const float* __restrict__ ln_scale,
                                const float* __restrict__ ln_bias,
                                float* __restrict__ out) {
    __shared__ float sh[256];
    int row = blockIdx.x;                  // b*S + s
    const float* hr = hidden + (size_t)row * cH;
    const float* cl = cache + ((size_t)(cW - 1) * cB * cS + row) * cH;  // cache[W-1][b][s]
    const float* ar = attn_out + (size_t)row * cH;

    float vals[4];
    float ls = 0.f;
#pragma unroll
    for (int t = 0; t < 4; ++t) {
        int i = threadIdx.x + t * 256;
        float g = 1.0f / (1.0f + __expf(-gate[i]));
        float comb = g * ar[i] + (1.0f - g) * cl[i];
        float o = hr[i] + lscale[i] * comb;
        vals[t] = o;
        ls += o;
        out[(size_t)row * cH + i] = o;
    }
    float mu = block_sum256(ls, sh) * (1.0f / cH);
    float lv = 0.f;
#pragma unroll
    for (int t = 0; t < 4; ++t) { float d = vals[t] - mu; lv += d * d; }
    float var = block_sum256(lv, sh) * (1.0f / cH);
    float inv = rsqrtf(var + cEPS);

    // new_cache slot W-1 region of d_out
    float* ns = out + (size_t)cB * cS * cH + ((size_t)(cW - 1) * cB * cS + row) * cH;
#pragma unroll
    for (int t = 0; t < 4; ++t) {
        int i = threadIdx.x + t * 256;
        ns[i] = (vals[t] - mu) * inv * ln_scale[i] + ln_bias[i];
    }
}

// ---------------------------------------------------------------------------
// Host launcher
// ---------------------------------------------------------------------------
extern "C" void kernel_launch(void* const* d_in, const int* in_sizes, int n_in,
                              void* d_out, int out_size, void* d_ws, size_t ws_size,
                              hipStream_t stream) {
    const float* hidden = (const float*)d_in[0];
    const float* cache  = (const float*)d_in[1];
    const float* ln_s   = (const float*)d_in[2];
    const float* ln_b   = (const float*)d_in[3];
    const float* Wq     = (const float*)d_in[4];
    const float* Wk     = (const float*)d_in[5];
    const float* Wv     = (const float*)d_in[6];
    const float* Wo     = (const float*)d_in[7];
    const float* gate   = (const float*)d_in[8];
    const float* lsc    = (const float*)d_in[9];
    float* out = (float*)d_out;

    // workspace layout (bytes)
    char* ws = (char*)d_ws;
    size_t off = 0;
    auto take = [&](size_t bytes) { char* p = ws + off; off += (bytes + 255) & ~(size_t)255; return (void*)p; };
    _Float16* kv    = (_Float16*)take((size_t)cB * cKS * cH * 2);       // 20 MB
    _Float16* WqT   = (_Float16*)take((size_t)cH * cH * 2);
    _Float16* WkT   = (_Float16*)take((size_t)cH * cH * 2);
    _Float16* WvT   = (_Float16*)take((size_t)cH * cH * 2);
    _Float16* WoT   = (_Float16*)take((size_t)cH * cH * 2);
    float*    qout  = (float*)take((size_t)cB * cS * cH * 4);
    float*    kout  = (float*)take((size_t)cB * cKS * cH * 4);
    float*    vout  = (float*)take((size_t)cB * cKS * cH * 4);
    _Float16* Qf    = (_Float16*)take((size_t)cB * cS * cH * 2);
    _Float16* Kf    = (_Float16*)take((size_t)cB * cKS * cH * 2);
    _Float16* Vt    = (_Float16*)take((size_t)cB * cKS * cH * 2);
    _Float16* ctx   = (_Float16*)take((size_t)cB * cS * cH * 2);
    float*    aout  = (float*)take((size_t)cB * cS * cH * 4);

    const int TB = 256;

    // 1) LayerNorm of hidden into kv rows [b][4S+s]
    ln_hidden_kernel<<<cB * cS, TB, 0, stream>>>(hidden, ln_s, ln_b, kv);
    // 2) cache -> kv rows [b][w*S+s]
    cache_to_kv_kernel<<<(cW * cB * cS * (cH / TB)), TB, 0, stream>>>(cache, kv);
    // 3) weight transpose+convert
    wt_convert_kernel<<<(cH * cH) / TB, TB, 0, stream>>>(Wq, WqT);
    wt_convert_kernel<<<(cH * cH) / TB, TB, 0, stream>>>(Wk, WkT);
    wt_convert_kernel<<<(cH * cH) / TB, TB, 0, stream>>>(Wv, WvT);
    wt_convert_kernel<<<(cH * cH) / TB, TB, 0, stream>>>(Wo, WoT);

    // 4) QKV GEMMs (TDM-staged, 128x128 tiles)
    {   // Q: M = B*S rows taken from kv at [b*KS + 4S + s]
        int M = cB * cS, N = cH, K = cH;
        int blocks = (M / 128) * (N / 128);
        gemm_tdm_kernel<<<blocks, TB, 0, stream>>>(kv, WqT, qout, M, N, K, cS, cKS, cW * cS);
    }
    {   // K and V: M = B*KS, identity row map within kv
        int M = cB * cKS, N = cH, K = cH;
        int blocks = (M / 128) * (N / 128);
        gemm_tdm_kernel<<<blocks, TB, 0, stream>>>(kv, WkT, kout, M, N, K, cKS, cKS, 0);
        gemm_tdm_kernel<<<blocks, TB, 0, stream>>>(kv, WvT, vout, M, N, K, cKS, cKS, 0);
    }

    // 5) RoPE + relayout
    rope_q_kernel<<<(cB * cS * cNH * 32) / TB, TB, 0, stream>>>(qout, Qf);
    rope_k_kernel<<<(cB * cKS * cNH * 32) / TB, TB, 0, stream>>>(kout, Kf);
    v_transpose_kernel<<<(cB * cKS * cH) / TB, TB, 0, stream>>>(vout, Vt);

    // 6) flash attention: B*NH*(S/16) waves / 8 per block
    attn_kernel<<<(cB * cNH * (cS / 16)) / 8, TB, 0, stream>>>(Qf, Kf, Vt, ctx);

    // 7) output projection: attn_out = ctx @ Wo
    {
        int M = cB * cS, N = cH, K = cH;
        int blocks = (M / 128) * (N / 128);
        gemm_tdm_kernel<<<blocks, TB, 0, stream>>>(ctx, WoT, aout, M, N, K, M, M, 0);
    }

    // 8) shift old cache slots 1..W-1 into new_cache slots 0..W-2
    hipMemcpyAsync(out + (size_t)cB * cS * cH,
                   cache + (size_t)cB * cS * cH,
                   (size_t)(cW - 1) * cB * cS * cH * sizeof(float),
                   hipMemcpyDeviceToDevice, stream);

    // 9) epilogue: gated residual + layer scale + LayerNorm -> new slot W-1
    epilogue_kernel<<<cB * cS, TB, 0, stream>>>(hidden, cache, aout, gate, lsc,
                                                ln_s, ln_b, out);
}
